// HyperAttention_85633057947785
// MI455X (gfx1250) — matrix operand order repaired and checked by reading, not compiled, not run on hardware
//
#include <hip/hip_runtime.h>
#include <math.h>

// ---------------- problem constants ----------------
#define Bq   2
#define Sq   8192
#define Hq   16
#define Dq   64
#define Pq   8
#define BSq  256
#define SAMPq 256
#define NBq  (Sq / BSq)          // 32
#define LOG_RATIO 3.4657359028f  // log(S/SAMP) = log(32)

typedef _Float16 f16;
typedef __attribute__((ext_vector_type(16))) _Float16 v16h;
typedef __attribute__((ext_vector_type(8)))  _Float16 v8h;
typedef __attribute__((ext_vector_type(8)))  float    v8f;

// ---------------- WMMA wrapper ----------------
__device__ __forceinline__ v8f wmma16(v16h a, v16h b, v8f c) {
    return __builtin_amdgcn_wmma_f32_16x16x32_f16(
        /*neg_a=*/false, a, /*neg_b=*/false, b,
        /*c_mod=*/(short)0, c, /*reuse_a=*/false, /*reuse_b=*/false);
}

// ---------------- fused DPP16 row_xmask butterfly steps (1 VALU op each) --
// dst = max(dpp_xmask(src), src) / dst = dpp_xmask(src) + src
#define ROW_XOR_MAX(x, MASK)                                                  \
    ({ float _r;                                                              \
       asm("v_max_num_f32_dpp %0, %1, %2 row_xmask:" #MASK                    \
           " row_mask:0xf bank_mask:0xf bound_ctrl:1"                         \
           : "=v"(_r) : "v"(x), "v"(x));                                      \
       _r; })
#define ROW_XOR_ADD(x, MASK)                                                  \
    ({ float _r;                                                              \
       asm("v_add_f32_dpp %0, %1, %2 row_xmask:" #MASK                        \
           " row_mask:0xf bank_mask:0xf bound_ctrl:1"                         \
           : "=v"(_r) : "v"(x), "v"(x));                                      \
       _r; })

__device__ __forceinline__ float red16_max(float x) {
    x = ROW_XOR_MAX(x, 1);
    x = ROW_XOR_MAX(x, 2);
    x = ROW_XOR_MAX(x, 4);
    x = ROW_XOR_MAX(x, 8);
    return x;
}
__device__ __forceinline__ float red16_sum(float x) {
    x = ROW_XOR_ADD(x, 1);
    x = ROW_XOR_ADD(x, 2);
    x = ROW_XOR_ADD(x, 4);
    x = ROW_XOR_ADD(x, 8);
    return x;
}

// A fragment: M=16 x K=32, source row-major [m][k], leading dim ld.
// ISA 7.12.2 16-bit A layout. Per lane the 16 halves are two contiguous
// 16-byte runs: [half*8 .. half*8+7] and [16+half*8 .. 16+half*8+7].
__device__ __forceinline__ v16h load_a_frag(const _Float16* base, int ld,
                                            int m0, int k0, int lane) {
    const int half = (lane >> 4) & 1;
    const int mr   = lane & 15;
    const _Float16* rowp = base + (m0 + mr) * ld + k0 + half * 8;
    v8h lo = *(const v8h*)(rowp);           // ds_load_b128
    v8h hi = *(const v8h*)(rowp + 16);      // ds_load_b128
    return __builtin_shufflevector(lo, hi, 0, 1, 2, 3, 4, 5, 6, 7,
                                   8, 9, 10, 11, 12, 13, 14, 15);
}

// B fragment: K=32 x N=16, source stored as [n][k] row-major (B^T), ld.
// Same K order as A: per lane two 16-byte runs [half*8 .. half*8+7] and
// [16+half*8 .. 16+half*8+7].
__device__ __forceinline__ v16h load_b_frag(const _Float16* base, int ld,
                                            int n0, int k0, int lane) {
    const int half = (lane >> 4) & 1;
    const int nr   = lane & 15;
    const _Float16* rowp = base + (n0 + nr) * ld + k0 + half * 8;
    v8h lo = *(const v8h*)(rowp);           // ds_load_b128
    v8h hi = *(const v8h*)(rowp + 16);      // ds_load_b128
    return __builtin_shufflevector(lo, hi, 0, 1, 2, 3, 4, 5, 6, 7,
                                   8, 9, 10, 11, 12, 13, 14, 15);
}

// ---------------- LSH hash kernel ----------------
__global__ __launch_bounds__(256) void lsh_hash_kernel(
    const float* __restrict__ x, const float* __restrict__ proj,
    unsigned char* __restrict__ hash) {
    __shared__ float sproj[Dq * Pq];
    int t = threadIdx.x;
    for (int i = t; i < Dq * Pq; i += 256) sproj[i] = proj[i];
    __syncthreads();

    int gid = blockIdx.x * 256 + t;                 // over B*S*H
    const float* row = x + (size_t)gid * Dq;
    float acc[Pq];
#pragma unroll
    for (int p = 0; p < Pq; ++p) acc[p] = 0.0f;
    const float4* r4 = (const float4*)row;
#pragma unroll 4
    for (int dq = 0; dq < Dq / 4; ++dq) {
        float4 xv = r4[dq];
        const float* pr = &sproj[dq * 4 * Pq];
#pragma unroll
        for (int p = 0; p < Pq; ++p) {
            acc[p] += xv.x * pr[p] + xv.y * pr[Pq + p] +
                      xv.z * pr[2 * Pq + p] + xv.w * pr[3 * Pq + p];
        }
    }
    int bin = 0;
#pragma unroll
    for (int p = 0; p < Pq; ++p) bin |= (acc[p] > 0.0f) ? (1 << p) : 0;
    int gray = bin ^ (bin >> 1);                    // == _gray_perm[bin]
    int h = gid % Hq;
    int s = (gid / Hq) % Sq;
    int b = gid / (Hq * Sq);
    hash[((size_t)(b * Hq + h)) * Sq + s] = (unsigned char)gray;
}

// ---------------- stable counting sort per (b,h) ----------------
__global__ __launch_bounds__(256) void sort_kernel(
    const unsigned char* __restrict__ hash, int* __restrict__ sortidx) {
    __shared__ unsigned char sh[Sq];
    __shared__ unsigned int histo[256];
    __shared__ unsigned int offs[256];
    int bh = blockIdx.x;
    int t  = threadIdx.x;
    for (int i = t; i < Sq; i += 256) sh[i] = hash[(size_t)bh * Sq + i];
    histo[t] = 0;
    __syncthreads();
    for (int i = t; i < Sq; i += 256) atomicAdd(&histo[sh[i]], 1u);
    __syncthreads();
    if (t == 0) {
        unsigned int run = 0;
        for (int i = 0; i < 256; ++i) { offs[i] = run; run += histo[i]; }
    }
    __syncthreads();
    unsigned int pos = offs[t];                     // thread t owns bucket t
    for (int s = 0; s < Sq; ++s) {                  // broadcast LDS reads
        if (sh[s] == (unsigned char)t) sortidx[(size_t)bh * Sq + pos++] = s;
    }
}

// ---------------- flash attention pass (256 keys, online softmax) ---------
struct StripState {
    v8f   acc[4];     // 16 rows x 64 cols, unnormalized
    float m[8];       // running max (row = r + 8*half of lane)
    float l[8];       // running sum
};

__device__ __forceinline__ void flash_pass(
    const _Float16* sQ, const _Float16* sK, const _Float16* sVT,
    _Float16* sPw, int m0, int lane, float scale, StripState& st) {
    const int half = (lane >> 4) & 1;
    const int nr   = lane & 15;
    v16h aq0 = load_a_frag(sQ, Dq, m0, 0,  lane);
    v16h aq1 = load_a_frag(sQ, Dq, m0, 32, lane);
#pragma unroll
    for (int t2 = 0; t2 < 4; ++t2) st.acc[t2] = (v8f){};
#pragma unroll
    for (int r = 0; r < 8; ++r) { st.m[r] = -INFINITY; st.l[r] = 0.0f; }

    for (int kc = 0; kc < 8; ++kc) {                // 8 chunks of 32 keys
        const int n0 = kc * 32;
        v8f c0 = (v8f){}, c1 = (v8f){};
        v16h bf;
        bf = load_b_frag(sK, Dq, n0,      0,  lane); c0 = wmma16(aq0, bf, c0);
        bf = load_b_frag(sK, Dq, n0,      32, lane); c0 = wmma16(aq1, bf, c0);
        bf = load_b_frag(sK, Dq, n0 + 16, 0,  lane); c1 = wmma16(aq0, bf, c1);
        bf = load_b_frag(sK, Dq, n0 + 16, 32, lane); c1 = wmma16(aq1, bf, c1);

        float corr[8];
#pragma unroll
        for (int r = 0; r < 8; ++r) {
            float s0 = c0[r] * scale;
            float s1 = c1[r] * scale;
            float mx = red16_max(fmaxf(s0, s1));    // fused DPP16 butterfly
            float mn = fmaxf(st.m[r], mx);
            float cr = __expf(st.m[r] - mn);
            float p0 = __expf(s0 - mn);
            float p1 = __expf(s1 - mn);
            float rs = red16_sum(p0 + p1);          // fused DPP16 butterfly
            st.l[r] = st.l[r] * cr + rs;
            st.m[r] = mn;
            corr[r] = cr;
            int row = r + 8 * half;
            sPw[row * 32 + nr]      = (_Float16)p0;
            sPw[row * 32 + 16 + nr] = (_Float16)p1;
        }
        asm volatile("" ::: "memory");              // keep LDS store->load order
        v16h pa = load_a_frag(sPw, 32, 0, 0, lane);
#pragma unroll
        for (int t2 = 0; t2 < 4; ++t2) {
#pragma unroll
            for (int r = 0; r < 8; ++r) st.acc[t2][r] *= corr[r];
            v16h bv = load_b_frag(sVT, BSq, t2 * 16, n0, lane);
            st.acc[t2] = wmma16(pa, bv, st.acc[t2]);
        }
    }
}

// ---------------- fused block + residual attention ----------------
// grid = (NB, H, B), block = 256 (8 waves). Dynamic LDS layout:
//   sQ  [256*64] f16 (32KB) | sK [256*64] f16 (32KB) | sVT [64*256] f16 (32KB)
//   sP  [8][16*32] f16 (8KB) | sOrig [256] int (1KB)
__global__ __launch_bounds__(256) void hyper_attn_kernel(
    const float* __restrict__ q, const float* __restrict__ k,
    const float* __restrict__ v, const int* __restrict__ q_sort,
    const int* __restrict__ k_sort, const int* __restrict__ sample_idx,
    float* __restrict__ out) {
    extern __shared__ char smem[];
    _Float16* sQ  = (_Float16*)(smem);
    _Float16* sK  = (_Float16*)(smem + 32768);
    _Float16* sVT = (_Float16*)(smem + 65536);
    _Float16* sP  = (_Float16*)(smem + 98304);
    int*      sOrig = (int*)(smem + 106496);

    const int blk = blockIdx.x, h = blockIdx.y, b = blockIdx.z;
    const int t = threadIdx.x;
    const int wave = t >> 5, lane = t & 31;
    const int half = (lane >> 4) & 1, nr = lane & 15;
    const size_t bh = (size_t)(b * Hq + h);
    const float scale = 0.125f;                     // 1/sqrt(64)
    _Float16* sPw = sP + wave * (16 * 32);

    // ---- gather sorted Q / K / V tiles into LDS ----
    {
        int sq = q_sort[bh * Sq + (size_t)blk * BSq + t];
        sOrig[t] = sq;
        const float4* qr = (const float4*)(q + (((size_t)b * Sq + sq) * Hq + h) * Dq);
        int sk = k_sort[bh * Sq + (size_t)blk * BSq + t];
        const float4* kr = (const float4*)(k + (((size_t)b * Sq + sk) * Hq + h) * Dq);
        const float4* vr = (const float4*)(v + (((size_t)b * Sq + sk) * Hq + h) * Dq);
#pragma unroll
        for (int dq = 0; dq < Dq / 4; ++dq) {
            float4 xq = qr[dq], xk = kr[dq], xv = vr[dq];
            int d0 = dq * 4;
            sQ[t * Dq + d0]     = (_Float16)xq.x; sQ[t * Dq + d0 + 1] = (_Float16)xq.y;
            sQ[t * Dq + d0 + 2] = (_Float16)xq.z; sQ[t * Dq + d0 + 3] = (_Float16)xq.w;
            sK[t * Dq + d0]     = (_Float16)xk.x; sK[t * Dq + d0 + 1] = (_Float16)xk.y;
            sK[t * Dq + d0 + 2] = (_Float16)xk.z; sK[t * Dq + d0 + 3] = (_Float16)xk.w;
            sVT[(d0)     * BSq + t] = (_Float16)xv.x;   // V stored transposed
            sVT[(d0 + 1) * BSq + t] = (_Float16)xv.y;
            sVT[(d0 + 2) * BSq + t] = (_Float16)xv.z;
            sVT[(d0 + 3) * BSq + t] = (_Float16)xv.w;
        }
    }
    __syncthreads();

    // ---- phase 0: block-diagonal attention (both 16-row strips) ----
    StripState st0, st1;
    flash_pass(sQ, sK, sVT, sPw, wave * 32,      lane, scale, st0);
    flash_pass(sQ, sK, sVT, sPw, wave * 32 + 16, lane, scale, st1);
    __syncthreads();

    // ---- reload sampled K / V for residual phase ----
    {
        int sr = sample_idx[bh * SAMPq + t];
        const float4* kr = (const float4*)(k + (((size_t)b * Sq + sr) * Hq + h) * Dq);
        const float4* vr = (const float4*)(v + (((size_t)b * Sq + sr) * Hq + h) * Dq);
#pragma unroll
        for (int dq = 0; dq < Dq / 4; ++dq) {
            float4 xk = kr[dq], xv = vr[dq];
            int d0 = dq * 4;
            sK[t * Dq + d0]     = (_Float16)xk.x; sK[t * Dq + d0 + 1] = (_Float16)xk.y;
            sK[t * Dq + d0 + 2] = (_Float16)xk.z; sK[t * Dq + d0 + 3] = (_Float16)xk.w;
            sVT[(d0)     * BSq + t] = (_Float16)xv.x;
            sVT[(d0 + 1) * BSq + t] = (_Float16)xv.y;
            sVT[(d0 + 2) * BSq + t] = (_Float16)xv.z;
            sVT[(d0 + 3) * BSq + t] = (_Float16)xv.w;
        }
    }
    __syncthreads();

    // ---- phase 1: residual attention + logsumexp combine + scatter out ----
    for (int ss = 0; ss < 2; ++ss) {
        const int m0 = wave * 32 + ss * 16;
        StripState s2;
        flash_pass(sQ, sK, sVT, sPw, m0, lane, scale, s2);
        StripState& sb = ss ? st1 : st0;
#pragma unroll
        for (int r = 0; r < 8; ++r) {
            float lse1 = sb.m[r] + __logf(sb.l[r]);
            float lse2 = s2.m[r] + __logf(s2.l[r]) + LOG_RATIO;
            float mm   = fmaxf(lse1, lse2);
            float lseT = mm + __logf(__expf(lse1 - mm) + __expf(lse2 - mm));
            float e1 = __expf(sb.m[r] - lseT);               // = w_blk / l1
            float e2 = __expf(s2.m[r] + LOG_RATIO - lseT);   // = w_r   / l2
            int row = m0 + r + 8 * half;
            int so  = sOrig[row];
            float* orow = out + (((size_t)b * Sq + so) * Hq + h) * Dq + nr;
#pragma unroll
            for (int t2 = 0; t2 < 4; ++t2)
                orow[t2 * 16] = sb.acc[t2][r] * e1 + s2.acc[t2][r] * e2;
        }
    }
}

// ---------------- host launcher ----------------
extern "C" void kernel_launch(void* const* d_in, const int* in_sizes, int n_in,
                              void* d_out, int out_size, void* d_ws, size_t ws_size,
                              hipStream_t stream) {
    const float* q    = (const float*)d_in[0];
    const float* k    = (const float*)d_in[1];
    const float* v    = (const float*)d_in[2];
    const float* proj = (const float*)d_in[3];
    const int*   sidx = (const int*)d_in[4];
    float* out = (float*)d_out;

    char* ws = (char*)d_ws;
    unsigned char* q_hash = (unsigned char*)(ws);                    // 256 KB
    unsigned char* k_hash = (unsigned char*)(ws + 262144);           // 256 KB
    int* q_sort = (int*)(ws + 524288);                               // 1 MB
    int* k_sort = (int*)(ws + 524288 + 1048576);                     // 1 MB

    const int hash_blocks = (Bq * Sq * Hq) / 256;                    // 1024
    lsh_hash_kernel<<<hash_blocks, 256, 0, stream>>>(q, proj, q_hash);
    lsh_hash_kernel<<<hash_blocks, 256, 0, stream>>>(k, proj, k_hash);

    sort_kernel<<<Bq * Hq, 256, 0, stream>>>(q_hash, q_sort);
    sort_kernel<<<Bq * Hq, 256, 0, stream>>>(k_hash, k_sort);

    dim3 grid(NBq, Hq, Bq);
    size_t lds_bytes = 106496 + 1024;                                // ~105 KB
    hyper_attn_kernel<<<grid, 256, lds_bytes, stream>>>(
        q, k, v, q_sort, k_sort, sidx, out);
}
